// SelfAttention_6004364280374
// MI455X (gfx1250) — compile-verified, matrix operand order checked by reading, not run on hardware
//
#include <hip/hip_runtime.h>
#include <hip/hip_bf16.h>

// ---------------------------------------------------------------------------
// Causal MHA forward for MI455X (gfx1250, wave32).
// bf16 WMMA (f32 accum) everywhere; compute-bound per roofline (23.3 TB/s HBM
// moves ~150MB in ~6us; matrix-core work dominates). Data staging uses
// GLOBAL_LOAD_ASYNC_TO_LDS_B128 and the Tensor Data Mover where the toolchain
// exposes them (guarded; falls back to load+ds_store otherwise).
// ---------------------------------------------------------------------------

#define E_DIM 1024
#define SEQ   2048
#define NH    16
#define HD2   64
#define BATCH 4

typedef __attribute__((ext_vector_type(16))) __bf16 v16bf;
typedef __attribute__((ext_vector_type(8)))  float  v8f;

#if defined(__has_builtin)
#if __has_builtin(__builtin_amdgcn_global_load_async_to_lds_b128)
#define HAVE_ASYNC 1
#endif
#if __has_builtin(__builtin_amdgcn_tensor_load_to_lds)
#define HAVE_TDM 1
#endif
#endif

#ifdef HAVE_ASYNC
// Builtin signature (from hipcc diagnostic): param0 is a non-const pointer to
// a 16B int vector in the global AS; param1 the LDS-side equivalent.
typedef __attribute__((__vector_size__(16))) int i32x4_t;
typedef __attribute__((address_space(1))) i32x4_t* glb_v4i_ptr;
typedef __attribute__((address_space(3))) i32x4_t* lds_v4i_ptr;

// Per-lane async copy of 16B global -> LDS (tracked by ASYNCcnt).
__device__ inline void async_ld_b128(const void* g, void* l) {
    __builtin_amdgcn_global_load_async_to_lds_b128(
        (glb_v4i_ptr)(uintptr_t)g,
        (lds_v4i_ptr)(unsigned)(uintptr_t)l,
        0, 0);
}
__device__ inline void wait_async0() {
#if __has_builtin(__builtin_amdgcn_s_wait_asynccnt)
    __builtin_amdgcn_s_wait_asynccnt(0);
#else
    asm volatile("s_wait_asynccnt 0x0" ::: "memory");
#endif
}
#endif

#ifdef HAVE_TDM
typedef __attribute__((ext_vector_type(4))) unsigned int v4u_t;
typedef __attribute__((ext_vector_type(8))) int v8i_t;
typedef __attribute__((ext_vector_type(4))) int v4i_t;
// 2D TDM load: tile_d1 rows x tile_d0 elems (2B each), row stride
// row_stride elems in memory; optional LDS padding (pad codes per D# spec).
__device__ inline void tdm_load_2d(unsigned lds_byte, const void* gptr,
                                   int tile_d0, int tile_d1,
                                   int tensor_d0, int tensor_d1,
                                   int row_stride, int pad_interval_code,
                                   int pad_amount_code, int pad_en) {
    unsigned long long ga = (unsigned long long)(uintptr_t)gptr;
    v4u_t g0;
    g0.x = 1u;                                     // count=1, user descriptor
    g0.y = lds_byte;                               // lds_addr
    g0.z = (unsigned)(ga & 0xFFFFFFFFull);         // global_addr lo
    g0.w = (unsigned)((ga >> 32) & 0x01FFFFFFull) | (2u << 30); // hi + type=2
    v8i_t g1;
    g1[0] = (int)((1u << 16)                       // data_size = 2B
                | ((unsigned)pad_en << 20)
                | ((unsigned)pad_interval_code << 22)
                | ((unsigned)pad_amount_code << 25));
    g1[1] = (int)(((unsigned)tensor_d0 & 0xFFFFu) << 16);          // dim0 lo16
    g1[2] = (int)((((unsigned)tensor_d0 >> 16) & 0xFFFFu)
                | (((unsigned)tensor_d1 & 0xFFFFu) << 16));        // dim0 hi / dim1 lo
    g1[3] = (int)((((unsigned)tensor_d1 >> 16) & 0xFFFFu)
                | (((unsigned)tile_d0 & 0xFFFFu) << 16));          // dim1 hi / tile0
    g1[4] = (int)((unsigned)tile_d1 & 0xFFFFu);                    // tile1 (tile2=0)
    g1[5] = row_stride;                                            // dim0_stride lo32
    g1[6] = 0;
    g1[7] = 0;
    v4i_t z4 = {0, 0, 0, 0};
#if __clang_major__ >= 23
    v8i_t z8 = {0, 0, 0, 0, 0, 0, 0, 0};
    __builtin_amdgcn_tensor_load_to_lds(g0, g1, z4, z4, z8, 0);
#else
    __builtin_amdgcn_tensor_load_to_lds(g0, g1, z4, z4, 0);
#endif
}
#endif

// Load one 16x32 bf16 A/B fragment from LDS (CDNA5 16-bit A layout: lanes
// 0-15 hold K={0..7,16..23}, lanes 16-31 hold K={8..15,24..31} of row
// lane&15). Two ds_load_b128 per lane.
__device__ inline v16bf load_frag_lds(const __bf16* base, int stride, int lane) {
    const int m  = lane & 15;
    const int hi = lane >> 4;
    const __bf16* p = base + m * stride + hi * 8;
    union { v16bf v; uint4 q[2]; } u;
    u.q[0] = *reinterpret_cast<const uint4*>(p);
    u.q[1] = *reinterpret_cast<const uint4*>(p + 16);
    return u.v;
}

__device__ inline v8f wmma_bf16(v16bf a, v16bf b, v8f c) {
    return __builtin_amdgcn_wmma_f32_16x16x32_bf16(
        false, a, false, b, (short)0, c, false, false);
}

// ---------------------------------------------------------------------------
__global__ void cvt_f32_bf16(const float* __restrict__ src,
                             __bf16* __restrict__ dst, int n) {
    int i = blockIdx.x * blockDim.x + threadIdx.x;
    int stride = gridDim.x * blockDim.x;
    for (; i < n; i += stride) dst[i] = (__bf16)src[i];
}

// ---------------------------------------------------------------------------
// y = A @ W^T  (A: Mx1024 bf16, W: 1024x1024 bf16, both row-major).
// mode 0: bf16 head-major [B,NH,SEQ,HD2], *scale. mode 1: fp32 [M,1024]+bias.
// Block tile 128x128, BK=32; 8 waves each own 32x64 (2x4 WMMA / k-step).
// ---------------------------------------------------------------------------
__global__ __launch_bounds__(256) void gemm_xwt(
    const __bf16* __restrict__ A, const __bf16* __restrict__ W,
    __bf16* __restrict__ outB, float* __restrict__ outF,
    const float* __restrict__ bias, int mode, float scale)
{
    constexpr int BM = 128, BN = 128, BK = 32, LDA = 40; // 80B rows, 16B mult
    __shared__ __bf16 As[BM * LDA];
    __shared__ __bf16 Bs[BN * LDA];

    const int tid  = threadIdx.x;
    const int lane = tid & 31;
    const int w    = tid >> 5;
    const int wm   = (w & 3) * 32;
    const int wn   = (w >> 2) * 64;
    const int r0   = blockIdx.x * BM;
    const int c0   = blockIdx.y * BN;

    v8f acc[2][4] = {};

    for (int k0 = 0; k0 < E_DIM; k0 += BK) {
        // Stage A,B tiles (each 128x32): thread moves 16+16 contiguous bf16.
        const int r = tid >> 1, c = (tid & 1) * 16;
        const __bf16* ga = A + (size_t)(r0 + r) * E_DIM + k0 + c;
        const __bf16* gb = W + (size_t)(c0 + r) * E_DIM + k0 + c;
        __bf16* la = As + r * LDA + c;
        __bf16* lb = Bs + r * LDA + c;
#ifdef HAVE_ASYNC
        async_ld_b128(ga,     la);
        async_ld_b128(ga + 8, la + 8);
        async_ld_b128(gb,     lb);
        async_ld_b128(gb + 8, lb + 8);
        wait_async0();           // barrier does not cover ASYNCcnt
#else
        {
            const uint4* pa = reinterpret_cast<const uint4*>(ga);
            uint4 a0 = pa[0], a1 = pa[1];
            const uint4* pb = reinterpret_cast<const uint4*>(gb);
            uint4 b0 = pb[0], b1 = pb[1];
            uint4* sa = reinterpret_cast<uint4*>(la);
            sa[0] = a0; sa[1] = a1;
            uint4* sb = reinterpret_cast<uint4*>(lb);
            sb[0] = b0; sb[1] = b1;
        }
#endif
        __syncthreads();

        v16bf af0 = load_frag_lds(As + (wm +  0) * LDA, LDA, lane);
        v16bf af1 = load_frag_lds(As + (wm + 16) * LDA, LDA, lane);
        for (int tn = 0; tn < 4; ++tn) {
            v16bf bfr = load_frag_lds(Bs + (wn + tn * 16) * LDA, LDA, lane);
            acc[0][tn] = wmma_bf16(af0, bfr, acc[0][tn]);
            acc[1][tn] = wmma_bf16(af1, bfr, acc[1][tn]);
        }
        __syncthreads();
    }

    const int nl = lane & 15, hi = lane >> 4;
    for (int tm = 0; tm < 2; ++tm)
        for (int tn = 0; tn < 4; ++tn)
            for (int i = 0; i < 8; ++i) {
                int rr = r0 + wm + tm * 16 + i + 8 * hi;
                int cc = c0 + wn + tn * 16 + nl;
                float v = acc[tm][tn][i] * scale;
                if (mode == 0) {
                    int b = rr / SEQ, s = rr % SEQ;
                    int h = cc / HD2, d = cc % HD2;
                    outB[(((size_t)(b * NH + h)) * SEQ + s) * HD2 + d] = (__bf16)v;
                } else {
                    outF[(size_t)rr * E_DIM + cc] = v + bias[cc];
                }
            }
}

// ---------------------------------------------------------------------------
// Causal flash attention. One block = one (b,h) x 128-query stripe; 8 waves,
// each owning a 16-row slab. Q pre-scaled by 1/sqrt(hd).
// ---------------------------------------------------------------------------
__global__ __launch_bounds__(256) void attn_flash(
    const __bf16* __restrict__ Qh, const __bf16* __restrict__ Kh,
    const __bf16* __restrict__ Vh, __bf16* __restrict__ Oout)
{
    constexpr int BR = 128, BC = 64, LDT = 72; // 144B rows, 16B mult
    __shared__ __bf16 Qs[BR * LDT];
    __shared__ __bf16 Ks[BC * LDT];
    __shared__ __bf16 Vt[HD2 * LDT];          // V transposed: Vt[d][k]
    __shared__ __bf16 Ps[8 * 16 * LDT];       // per-wave P slabs

    const int tid  = threadIdx.x;
    const int lane = tid & 31;
    const int w    = tid >> 5;
    const int nl   = lane & 15, hi = lane >> 4;

    const int gid = blockIdx.x;
    const int qb  = gid & 15;          // SEQ/BR = 16
    const int h   = (gid >> 4) & 15;
    const int b   = gid >> 8;
    const int qBase = qb * BR;

    const size_t headOff = ((size_t)(b * NH + h)) * SEQ * HD2;
    const __bf16* Qg = Qh + headOff;
    const __bf16* Kg = Kh + headOff;
    const __bf16* Vg = Vh + headOff;

#ifdef HAVE_TDM
    // TDM: 128x64 bf16 tile, hardware-padded to LDS stride 72 elems
    // (pad every 32 dwords (code 4) by 4 dwords (code 3)).
    if (w == 0) {
        tdm_load_2d((unsigned)(uintptr_t)&Qs[0], Qg + (size_t)qBase * HD2,
                    /*tile*/ HD2, BR, /*tensor*/ HD2, SEQ, /*stride*/ HD2,
                    /*pad_interval*/ 4, /*pad_amount*/ 3, /*pad_en*/ 1);
        __builtin_amdgcn_s_wait_tensorcnt((short)0);
    }
#else
    { // Q tile 128x64: each thread 32 contiguous bf16
        int r = tid >> 1, c = (tid & 1) * 32;
        const uint4* g = reinterpret_cast<const uint4*>(Qg + (size_t)(qBase + r) * HD2 + c);
        uint4* s = reinterpret_cast<uint4*>(Qs + r * LDT + c);
        s[0] = g[0]; s[1] = g[1]; s[2] = g[2]; s[3] = g[3];
    }
#endif

    float M[8], L[8];
    v8f Oacc[4] = {};
    for (int i = 0; i < 8; ++i) { M[i] = -__builtin_inff(); L[i] = 0.f; }

    __bf16* PsW = Ps + w * 16 * LDT;
    const int nkb = (qb + 1) * (BR / BC); // causal: kBase < qBase+BR

    for (int kb = 0; kb < nkb; ++kb) {
        const int kBase = kb * BC;
        __syncthreads(); // previous iteration's Ks/Vt consumers (+Q stage, 1st iter)

        { // K tile 64x64
            int r = tid >> 2, c = (tid & 3) * 16;
            const __bf16* gk = Kg + (size_t)(kBase + r) * HD2 + c;
            __bf16* lk = Ks + r * LDT + c;
#ifdef HAVE_ASYNC
            async_ld_b128(gk,     lk);
            async_ld_b128(gk + 8, lk + 8);
#else
            const uint4* g = reinterpret_cast<const uint4*>(gk);
            uint4* s = reinterpret_cast<uint4*>(lk);
            s[0] = g[0]; s[1] = g[1];
#endif
        }
        { // V tile 64x64, transposed into Vt[d][k]
            int r = tid >> 2, c = (tid & 3) * 16;
            union { uint4 q[2]; __bf16 e[16]; } u;
            const uint4* g = reinterpret_cast<const uint4*>(Vg + (size_t)(kBase + r) * HD2 + c);
            u.q[0] = g[0]; u.q[1] = g[1];
            for (int i = 0; i < 16; ++i) Vt[(c + i) * LDT + r] = u.e[i];
        }
#ifdef HAVE_ASYNC
        wait_async0();
#endif
        __syncthreads();

        // S = Q_slab(16x64) @ K^T: column j of B is row j of Ks (contiguous).
        v8f sacc[4] = {};
        for (int c = 0; c < HD2; c += 32) {
            v16bf af = load_frag_lds(Qs + (w * 16) * LDT + c, LDT, lane);
            for (int j = 0; j < 4; ++j) {
                v16bf bk = load_frag_lds(Ks + (j * 16) * LDT + c, LDT, lane);
                sacc[j] = wmma_bf16(af, bk, sacc[j]);
            }
        }

        // Causal mask + online softmax. C-layout rows i+8*hi live in 16-lane
        // halves; xor masks 1,2,4,8 reduce each half independently.
        float rm[8], rs[8], mnew[8], alpha[8];
        for (int i = 0; i < 8; ++i) {
            int qg = qBase + w * 16 + i + 8 * hi;
            float v = -__builtin_inff();
            for (int j = 0; j < 4; ++j) {
                int kg = kBase + j * 16 + nl;
                float sv = (kg <= qg) ? (float)sacc[j][i] : -__builtin_inff();
                sacc[j][i] = sv;
                v = fmaxf(v, sv);
            }
            rm[i] = v;
        }
        for (int m = 1; m < 16; m <<= 1)
            for (int i = 0; i < 8; ++i)
                rm[i] = fmaxf(rm[i], __shfl_xor(rm[i], m, 32));

        for (int i = 0; i < 8; ++i) {
            mnew[i]  = fmaxf(M[i], rm[i]);
            alpha[i] = __expf(M[i] - mnew[i]);
            rs[i]    = 0.f;
        }
        for (int j = 0; j < 4; ++j)
            for (int i = 0; i < 8; ++i) {
                float p = __expf((float)sacc[j][i] - mnew[i]);
                rs[i] += p;
                PsW[(i + 8 * hi) * LDT + j * 16 + nl] = (__bf16)p;
            }
        for (int m = 1; m < 16; m <<= 1)
            for (int i = 0; i < 8; ++i)
                rs[i] += __shfl_xor(rs[i], m, 32);
        for (int i = 0; i < 8; ++i) {
            L[i] = L[i] * alpha[i] + rs[i];
            M[i] = mnew[i];
        }
        for (int j = 0; j < 4; ++j)
            for (int i = 0; i < 8; ++i)
                Oacc[j][i] = (float)Oacc[j][i] * alpha[i];

        // O += P(16x64) @ V: column d of B is row d of Vt (contiguous).
        // PsW written+read by this wave only; same-wave LDS ops are ordered.
        for (int c = 0; c < BC; c += 32) {
            v16bf pa = load_frag_lds(PsW + c, LDT, lane);
            for (int j = 0; j < 4; ++j) {
                v16bf vb = load_frag_lds(Vt + (j * 16) * LDT + c, LDT, lane);
                Oacc[j] = wmma_bf16(pa, vb, Oacc[j]);
            }
        }
    }

    for (int j = 0; j < 4; ++j)
        for (int i = 0; i < 8; ++i) {
            int s = qBase + w * 16 + i + 8 * hi;
            int d = j * 16 + nl;
            float v = (float)Oacc[j][i] / L[i];
            Oout[((size_t)(b * SEQ + s)) * E_DIM + h * HD2 + d] = (__bf16)v;
        }
}

// ---------------------------------------------------------------------------
extern "C" void kernel_launch(void* const* d_in, const int* in_sizes, int n_in,
                              void* d_out, int out_size, void* d_ws, size_t ws_size,
                              hipStream_t stream) {
    (void)in_sizes; (void)n_in; (void)out_size; (void)ws_size;

    const float* x  = (const float*)d_in[0];
    const float* Wq = (const float*)d_in[1];
    const float* Wk = (const float*)d_in[2];
    const float* Wv = (const float*)d_in[3];
    const float* Wo = (const float*)d_in[4];
    const float* bo = (const float*)d_in[5];
    float* out = (float*)d_out;

    const int NX = BATCH * SEQ * E_DIM;   // 8388608
    const int NW = E_DIM * E_DIM;         // 1048576

    char* ws = (char*)d_ws;
    size_t off = 0;
    __bf16* xb  = (__bf16*)(ws + off); off += (size_t)NX * 2; // reused as attn_out
    __bf16* Wqb = (__bf16*)(ws + off); off += (size_t)NW * 2;
    __bf16* Wkb = (__bf16*)(ws + off); off += (size_t)NW * 2;
    __bf16* Wvb = (__bf16*)(ws + off); off += (size_t)NW * 2;
    __bf16* Wob = (__bf16*)(ws + off); off += (size_t)NW * 2;
    __bf16* Qh  = (__bf16*)(ws + off); off += (size_t)NX * 2;
    __bf16* Kh  = (__bf16*)(ws + off); off += (size_t)NX * 2;
    __bf16* Vh  = (__bf16*)(ws + off); off += (size_t)NX * 2;
    __bf16* AO  = xb; // x (bf16) dead once Q/K/V are built

    cvt_f32_bf16<<<4096, 256, 0, stream>>>(x,  xb,  NX);
    cvt_f32_bf16<<<1024, 256, 0, stream>>>(Wq, Wqb, NW);
    cvt_f32_bf16<<<1024, 256, 0, stream>>>(Wk, Wkb, NW);
    cvt_f32_bf16<<<1024, 256, 0, stream>>>(Wv, Wvb, NW);
    cvt_f32_bf16<<<1024, 256, 0, stream>>>(Wo, Wob, NW);

    dim3 ggrid(BATCH * SEQ / 128, E_DIM / 128);
    gemm_xwt<<<ggrid, 256, 0, stream>>>(xb, Wqb, Qh, nullptr, nullptr, 0, 0.125f);
    gemm_xwt<<<ggrid, 256, 0, stream>>>(xb, Wkb, Kh, nullptr, nullptr, 0, 1.0f);
    gemm_xwt<<<ggrid, 256, 0, stream>>>(xb, Wvb, Vh, nullptr, nullptr, 0, 1.0f);

    attn_flash<<<BATCH * NH * (SEQ / 128), 256, 0, stream>>>(Qh, Kh, Vh, AO);

    gemm_xwt<<<ggrid, 256, 0, stream>>>(AO, Wob, nullptr, out, bo, 1, 1.0f);
}
